// HyConv_24292335026750
// MI455X (gfx1250) — matrix operand order repaired and checked by reading, not compiled, not run on hardware
//
#include <hip/hip_runtime.h>
#include <hip/hip_bf16.h>

// Problem constants (match reference)
#define NB 4
#define VV 4096
#define EE 2048
#define CC 64   // C_IN == C_OUT == 64

typedef _Float16 f16;
typedef __attribute__((ext_vector_type(8)))  _Float16 v8h;
typedef __attribute__((ext_vector_type(16))) _Float16 v16h;
typedef __attribute__((ext_vector_type(8)))  float    v8f;
typedef __attribute__((ext_vector_type(4)))  float    v4f;

// ---------------------------------------------------------------------------
// Fragment loaders matching CDNA5 WMMA 16x16x32 f16 VGPR layouts (wave32).
// A (16x32, row m = lane&15, half = lane>>4):
//   element i -> K = half*8 + (i&7) + (i>>3)*16  => two contiguous 8-half chunks
// B (32x16, col n = lane&15, half = lane>>4):
//   element i -> K = half*8 + (i&7) + (i>>3)*16  => two contiguous 8-half chunks
// C/D (16x16): vgpr r -> M = r + 8*half, N = lane&15
// ---------------------------------------------------------------------------

__device__ __forceinline__ v16h load_a_f16(const f16* p /* row + k0 + half*8 */) {
    v8h lo = *(const v8h*)(p);
    v8h hi = *(const v8h*)(p + 16);
    return __builtin_shufflevector(lo, hi, 0,1,2,3,4,5,6,7,8,9,10,11,12,13,14,15);
}

__device__ __forceinline__ v16h load_a_f32(const float* p /* row + k0 + half*8 */) {
    const v4f* q = (const v4f*)p;
    v4f c0 = q[0], c1 = q[1], c2 = q[4], c3 = q[5];
    v16h a;
#pragma unroll
    for (int i = 0; i < 4; ++i) { a[i]    = (f16)c0[i]; a[4+i]  = (f16)c1[i]; }
#pragma unroll
    for (int i = 0; i < 4; ++i) { a[8+i]  = (f16)c2[i]; a[12+i] = (f16)c3[i]; }
    return a;
}

__device__ __forceinline__ v16h load_b_f16(const f16* p /* BT row + k0 + half*8 */) {
    v8h lo = *(const v8h*)(p);
    v8h hi = *(const v8h*)(p + 16);
    return __builtin_shufflevector(lo, hi, 0,1,2,3,4,5,6,7,8,9,10,11,12,13,14,15);
}

__device__ __forceinline__ v8f wmma32(v16h a, v16h b, v8f c) {
    return __builtin_amdgcn_wmma_f32_16x16x32_f16(false, a, false, b, (short)0, c, false, false);
}

// ---------------------------------------------------------------------------
// Single pass over f32 H: emit H_f16 [V,E] (straight) and Ht_f16 [E,V]
// (LDS-tiled transpose). H is 0/1 so f16 is exact.
// ---------------------------------------------------------------------------
__global__ void convert_transpose_h(const float* __restrict__ H,
                                    f16* __restrict__ Hf, f16* __restrict__ Ht) {
    __shared__ f16 tile[32][33];
    int e0 = blockIdx.x * 32, v0 = blockIdx.y * 32, n = blockIdx.z;
    const float* Hb  = H  + (size_t)n * VV * EE;
    f16*         Hfb = Hf + (size_t)n * VV * EE;
    f16*         Htb = Ht + (size_t)n * EE * VV;
    int tx = threadIdx.x & 31, ty = threadIdx.x >> 5;          // 32x8 threads
#pragma unroll
    for (int r = ty; r < 32; r += 8) {
        f16 h = (f16)Hb[(size_t)(v0 + r) * EE + e0 + tx];
        Hfb[(size_t)(v0 + r) * EE + e0 + tx] = h;              // coalesced f16 copy
        tile[r][tx] = h;
    }
    __syncthreads();
#pragma unroll
    for (int r = ty; r < 32; r += 8)
        Htb[(size_t)(e0 + r) * VV + v0 + tx] = tile[tx][r];    // coalesced transpose
}

// ---------------------------------------------------------------------------
// Generic wave-per-row f16 reduction -> reciprocal. Exact: summands are 0/1.
// dv = 1/rowsum(Hf [V,E]) ; de = 1/rowsum(Ht [E,V])
// ---------------------------------------------------------------------------
template <int ROWLEN>
__global__ void invdeg_rows_f16(const f16* __restrict__ M, float* __restrict__ inv) {
    int wid  = (blockIdx.x * blockDim.x + threadIdx.x) >> 5;   // one wave per row
    int lane = threadIdx.x & 31;
    const f16* row = M + (size_t)wid * ROWLEN;
    float s = 0.f;
#pragma unroll
    for (int i = lane * 8; i < ROWLEN; i += 32 * 8) {
        v8h v = *(const v8h*)(row + i);
#pragma unroll
        for (int j = 0; j < 8; ++j) s += (float)v[j];
    }
#pragma unroll
    for (int off = 16; off; off >>= 1) s += __shfl_xor(s, off, 32);
    if (lane == 0) inv[wid] = 1.0f / s;
}

__global__ void theta_transpose(const float* __restrict__ th, f16* __restrict__ thT) {
    int t = blockIdx.x * blockDim.x + threadIdx.x;             // 64*64 threads
    int c = t >> 6, o = t & 63;
    thT[o * CC + c] = (f16)th[c * CC + o];
}

// ---------------------------------------------------------------------------
// GEMM1: yT[n][o][v] = (x[n] @ theta)[v][o]    (A = x f32 cvt, B = thetaT)
// one wave -> 16 rows x 64 cols strip; K = 64 (tiny)
// ---------------------------------------------------------------------------
__global__ void gemm_x_theta(const float* __restrict__ x, const f16* __restrict__ thT,
                             f16* __restrict__ yT) {
    int wid  = (blockIdx.x * blockDim.x + threadIdx.x) >> 5;
    int lane = threadIdx.x & 31;
    const int strips = VV / 16;
    int n = wid / strips, v0 = (wid % strips) * 16;
    int half = lane >> 4, nlo = lane & 15;

    const float* A = x  + (size_t)n * VV * CC + (size_t)(v0 + nlo) * CC + half * 8;
    f16*         Y = yT + (size_t)n * CC * VV;

    v8f acc[4] = {};
#pragma unroll
    for (int k0 = 0; k0 < CC; k0 += 32) {
        v16h a = load_a_f32(A + k0);
#pragma unroll
        for (int t = 0; t < 4; ++t) {
            v16h b = load_b_f16(thT + (size_t)(16 * t + nlo) * CC + k0 + half * 8);
            acc[t] = wmma32(a, b, acc[t]);
        }
    }
#pragma unroll
    for (int t = 0; t < 4; ++t) {
        v8h h;
#pragma unroll
        for (int r = 0; r < 8; ++r) h[r] = (f16)acc[t][r];
        *(v8h*)(Y + (size_t)(16 * t + nlo) * VV + v0 + 8 * half) = h;
    }
}

// ---------------------------------------------------------------------------
// GEMM2: yET[n][o][e] = de[e] * (Ht[n] @ y)[e][o]   (A = Ht f16, B = yT)
// ---------------------------------------------------------------------------
__global__ void gemm_edge(const f16* __restrict__ Ht, const f16* __restrict__ yT,
                          const float* __restrict__ de, f16* __restrict__ yET) {
    int wid  = (blockIdx.x * blockDim.x + threadIdx.x) >> 5;
    int lane = threadIdx.x & 31;
    const int strips = EE / 16;
    int n = wid / strips, e0 = (wid % strips) * 16;
    int half = lane >> 4, nlo = lane & 15;

    const f16* A = Ht  + (size_t)n * EE * VV + (size_t)(e0 + nlo) * VV + half * 8;
    const f16* B = yT  + (size_t)n * CC * VV;
    f16*       Z = yET + (size_t)n * CC * EE;

    v8f acc[4] = {};
#pragma unroll 2
    for (int k0 = 0; k0 < VV; k0 += 32) {
        __builtin_prefetch(A + k0 + 8 * 32, 0, 0);             // A stream ahead
        v16h a = load_a_f16(A + k0);
#pragma unroll
        for (int t = 0; t < 4; ++t) {
            v16h b = load_b_f16(B + (size_t)(16 * t + nlo) * VV + k0 + half * 8);
            acc[t] = wmma32(a, b, acc[t]);
        }
    }
    const float* des = de + (size_t)n * EE + e0 + 8 * half;    // scale rows by de[e]
#pragma unroll
    for (int t = 0; t < 4; ++t) {
        v8h h;
#pragma unroll
        for (int r = 0; r < 8; ++r) h[r] = (f16)(acc[t][r] * des[r]);
        *(v8h*)(Z + (size_t)(16 * t + nlo) * EE + e0 + 8 * half) = h;
    }
}

// ---------------------------------------------------------------------------
// GEMM3: out[n][v][o] = dv[v] * (Hf[n] @ yE)[v][o] + bias[o]
// (A = Hf f16, B = yET) — pure f16 hot loop, no conversions
// ---------------------------------------------------------------------------
__global__ void gemm_vertex(const f16* __restrict__ Hf, const f16* __restrict__ yET,
                            const float* __restrict__ dv, const float* __restrict__ bias,
                            float* __restrict__ out) {
    int wid  = (blockIdx.x * blockDim.x + threadIdx.x) >> 5;
    int lane = threadIdx.x & 31;
    const int strips = VV / 16;
    int n = wid / strips, v0 = (wid % strips) * 16;
    int half = lane >> 4, nlo = lane & 15;

    const f16* A = Hf  + (size_t)n * VV * EE + (size_t)(v0 + nlo) * EE + half * 8;
    const f16* B = yET + (size_t)n * CC * EE;
    float*     O = out + (size_t)n * VV * CC;

    v8f acc[4] = {};
#pragma unroll 2
    for (int k0 = 0; k0 < EE; k0 += 32) {
        __builtin_prefetch(A + k0 + 8 * 32, 0, 0);             // A stream ahead
        v16h a = load_a_f16(A + k0);
#pragma unroll
        for (int t = 0; t < 4; ++t) {
            v16h b = load_b_f16(B + (size_t)(16 * t + nlo) * EE + k0 + half * 8);
            acc[t] = wmma32(a, b, acc[t]);
        }
    }
    const float* dvs = dv + (size_t)n * VV + v0 + 8 * half;
#pragma unroll
    for (int t = 0; t < 4; ++t) {
        int o = 16 * t + nlo;
        float bo = bias[o];
#pragma unroll
        for (int r = 0; r < 8; ++r) {
            int v = v0 + r + 8 * half;
            O[(size_t)v * CC + o] = acc[t][r] * dvs[r] + bo;
        }
    }
}

// ---------------------------------------------------------------------------
extern "C" void kernel_launch(void* const* d_in, const int* in_sizes, int n_in,
                              void* d_out, int out_size, void* d_ws, size_t ws_size,
                              hipStream_t stream) {
    const float* x     = (const float*)d_in[0];   // [N,V,C]
    const float* H     = (const float*)d_in[1];   // [N,V,E]
    const float* theta = (const float*)d_in[2];   // [C,O]
    const float* bias  = (const float*)d_in[3];   // [O]
    float*       out   = (float*)d_out;           // [N,V,O]

    // Workspace carve-up (~137 MB total)
    char* ws = (char*)d_ws;
    float* dv  = (float*)ws;   ws += (size_t)NB * VV * sizeof(float);
    float* de  = (float*)ws;   ws += (size_t)NB * EE * sizeof(float);
    f16*   thT = (f16*)ws;     ws += (size_t)CC * CC * sizeof(f16);
    f16*   yT  = (f16*)ws;     ws += (size_t)NB * CC * VV * sizeof(f16);
    f16*   yET = (f16*)ws;     ws += (size_t)NB * CC * EE * sizeof(f16);
    f16*   Hf  = (f16*)ws;     ws += (size_t)NB * VV * EE * sizeof(f16);
    f16*   Ht  = (f16*)ws;     // NB*EE*VV f16

    // 1) single f32 pass over H -> f16 copy + f16 transpose
    convert_transpose_h<<<dim3(EE / 32, VV / 32, NB), 256, 0, stream>>>(H, Hf, Ht);

    // 2) degrees from f16 copies (coalesced, mostly L2-resident)
    invdeg_rows_f16<EE><<<(NB * VV * 32) / 256, 256, 0, stream>>>(Hf, dv);
    invdeg_rows_f16<VV><<<(NB * EE * 32) / 256, 256, 0, stream>>>(Ht, de);

    // 3) operand prep + y = x @ theta
    theta_transpose<<<(CC * CC) / 256, 256, 0, stream>>>(theta, thT);
    gemm_x_theta<<<(NB * (VV / 16) * 32) / 256, 256, 0, stream>>>(x, thT, yT);

    // 4) yE = diag(de) Ht @ y    (512 waves, K = 4096)
    gemm_edge<<<(NB * (EE / 16) * 32) / 256, 256, 0, stream>>>(Ht, yT, de, yET);

    // 5) out = diag(dv) Hf @ yE + bias   (1024 waves, K = 2048)
    gemm_vertex<<<(NB * (VV / 16) * 32) / 256, 256, 0, stream>>>(Hf, yET, dv, bias, out);
}
